// VAEKawai_56770877719101
// MI455X (gfx1250) — matrix operand order repaired and checked by reading, not compiled, not run on hardware
//
#include <hip/hip_runtime.h>

// ---------------- problem constants ----------------
#define B_ 256
#define S_ 128
#define V_ 128
#define C_ 16
#define H_ 1024
#define Z_ 256
#define NSTEP_ 128
#define VC_ (V_ * C_)
#define H3_ (3 * H_)
#define KDEC_ (VC_ + Z_)   // 2304

typedef unsigned short u16;
typedef __bf16 bf16_t;
typedef bf16_t v16bf __attribute__((ext_vector_type(16)));
typedef float v8f __attribute__((ext_vector_type(8)));

struct alignas(16) U128 { unsigned int v[4]; };
struct alignas(32) U256 { U128 lo, hi; };

__device__ __forceinline__ u16 f32_to_bf16(float f) {
  unsigned int u = __float_as_uint(f);
  unsigned int r = u + 0x7FFFu + ((u >> 16) & 1u);   // round-to-nearest-even
  return (u16)(r >> 16);
}
__device__ __forceinline__ float sigm(float x) { return 1.0f / (1.0f + expf(-x)); }

// CDNA5 async global->LDS copy (ASYNCcnt path, no VGPR data movement).
// vdst = VGPR with 32-bit LDS byte address (low 32 bits of generic LDS ptr),
// vaddr = 64-bit global VA, saddr = off.  (ISA 15.18.3 op 98)
__device__ __forceinline__ void async_b128(const u16* g, u16* l) {
  unsigned lds_off = (unsigned)(unsigned long long)l;
  unsigned long long ga = (unsigned long long)g;
  asm volatile("global_load_async_to_lds_b128 %0, %1, off"
               :: "v"(lds_off), "v"(ga)
               : "memory");
}
__device__ __forceinline__ void wait_async0() {
  asm volatile("s_wait_asynccnt 0x0" ::: "memory");
}

// ---------------- WMMA GEMM: C[M,N] = A[M,K] * Wt[K,N] + bias ----------------
// A:  bf16 row-major (lda).
// Wt: bf16, PRE-TRANSPOSED weights, (K,N) row-major with row stride N.
// C:  f32 (ldc = N).
// grid = (N/128, M/64), block = 256 threads = 8 waves, async double-buffered LDS.
#define BM 64
#define BN 128
#define BK 32
#define APAD 8   // A row stride 40 u16 = 80B  (keeps ds_load_b128 16B-aligned)
#define BPAD 8   // B row stride 136 u16 = 272B

__global__ __launch_bounds__(256, 2)
void k_gemm_bf16(const u16* __restrict__ A, int lda,
                 const u16* __restrict__ Wt,
                 const float* __restrict__ bias,
                 float* __restrict__ C, int N, int K)
{
  __shared__ u16 As[2][BM][BK + APAD];   // [m][k]
  __shared__ u16 Bs[2][BK][BN + BPAD];   // [k][n]

  const int tid  = threadIdx.x;
  const int lane = tid & 31;
  const int wave = tid >> 5;
  const int wrow = wave & 3;    // M sub-tile (x16)
  const int wcol = wave >> 2;   // N sub-tile (x64)
  const int lm   = lane & 15;
  const int lh   = lane >> 4;

  const int nBlock = blockIdx.x * BN;
  const int mBlock = blockIdx.y * BM;

  // global->LDS staging indices (one b128 async copy each)
  const int am = tid >> 2;           // 0..63  (A row in tile)
  const int ak = (tid & 3) * 8;      // 0,8,16,24
  const int bk = tid >> 4;           // 0..15  (B k-row; +16 for second half)
  const int bn = (tid & 15) * 8;     // 0..120 (B n offset)

  const u16* aPtr = &A[(size_t)(mBlock + am) * lda + ak];
  const u16* bPtr = &Wt[(size_t)bk * N + nBlock + bn];
  const size_t bRow16 = (size_t)16 * N;
  const size_t bStep  = (size_t)BK * N;

  v8f acc[4] = {v8f{}, v8f{}, v8f{}, v8f{}};

  // ---- preload tile 0 (async to LDS) ----
  async_b128(aPtr,          &As[0][am][ak]);
  async_b128(bPtr,          &Bs[0][bk][bn]);
  async_b128(bPtr + bRow16, &Bs[0][bk + 16][bn]);
  wait_async0();
  __syncthreads();

  for (int k0 = 0, it = 0; k0 < K; k0 += BK, ++it) {
    const int  cur      = it & 1;
    const bool has_next = (k0 + BK) < K;

    // ---- kick off next tile's async copies (overlap with WMMA below) ----
    if (has_next) {
      const int nxt = cur ^ 1;
      const u16* bp = bPtr + (size_t)(it + 1) * bStep;
      async_b128(aPtr + (k0 + BK), &As[nxt][am][ak]);
      async_b128(bp,               &Bs[nxt][bk][bn]);
      async_b128(bp + bRow16,      &Bs[nxt][bk + 16][bn]);
    }
    if (k0 + 2 * BK < K) {   // prefetch two tiles ahead into L2
      __builtin_prefetch(aPtr + (k0 + 2 * BK), 0, 1);
      __builtin_prefetch(bPtr + (size_t)(it + 2) * bStep, 0, 1);
    }

    // ---- pull all fragments into registers first ----
    // A fragment: 16x32, lane lm = M row, lh selects K half
    U256 ar;
    {
      const u16* ap = &As[cur][wrow * 16 + lm][lh * 8];
      ar.lo = *reinterpret_cast<const U128*>(ap);        // K = lh*8 .. +7
      ar.hi = *reinterpret_cast<const U128*>(ap + 16);   // K = 16+lh*8 .. +7
    }
    // 4 B fragments (32x16 each): lane lm = N column, lh selects K half (K = lh*8.., 16+lh*8..)
    U256 br[4];
#pragma unroll
    for (int c = 0; c < 4; ++c) {
      const u16* bp = &Bs[cur][lh * 8][wcol * 64 + c * 16 + lm];
      u16 tb[16];
#pragma unroll
      for (int s = 0; s < 8; ++s) {
        tb[s]     = bp[s * (BN + BPAD)];
        tb[8 + s] = bp[(16 + s) * (BN + BPAD)];
      }
      __builtin_memcpy(&br[c], tb, sizeof(tb));
    }

    // ---- 4 back-to-back WMMAs ----
    v16bf afrag = __builtin_bit_cast(v16bf, ar);
#pragma unroll
    for (int c = 0; c < 4; ++c) {
      v16bf bfrag = __builtin_bit_cast(v16bf, br[c]);
      acc[c] = __builtin_amdgcn_wmma_f32_16x16x32_bf16(
          false, afrag, false, bfrag, (short)0, acc[c], false, false);
    }

    if (has_next) wait_async0();   // next tile resident before the barrier
    __syncthreads();
  }

  // ---- epilogue: D layout: M = r + 8*lh, N = lm ----
#pragma unroll
  for (int c = 0; c < 4; ++c) {
    const int col  = nBlock + wcol * 64 + c * 16 + lm;
    const int row0 = mBlock + wrow * 16 + lh * 8;
    const float bv = bias[col];
#pragma unroll
    for (int r = 0; r < 8; ++r)
      C[(size_t)(row0 + r) * N + col] = acc[c][r] + bv;
  }
}

// ---------------- elementwise kernels ----------------
// W (N,K) f32 -> Wt (K,N) bf16  (one-time; coalesced on output)
__global__ void k_cvt_w_t(const float* __restrict__ W, u16* __restrict__ Wt,
                          int N, int K) {
  int i = blockIdx.x * blockDim.x + threadIdx.x;   // i = k*N + n
  if (i >= N * K) return;
  int n = i % N;
  int k = i / N;
  Wt[i] = f32_to_bf16(W[(size_t)n * K + k]);
}

// x (B,V,S) f32 -> xt (S,B,V) bf16
__global__ void k_cvt_x(const float* __restrict__ x, u16* __restrict__ xt) {
  int i = blockIdx.x * blockDim.x + threadIdx.x;   // t*B*V + b*V + v
  int v = i % V_;
  int b = (i / V_) % B_;
  int t = i / (V_ * B_);
  xt[i] = f32_to_bf16(x[(size_t)b * V_ * S_ + (size_t)v * S_ + t]);
}

__global__ void k_zero_f32(float* __restrict__ p, int n) {
  int i = blockIdx.x * blockDim.x + threadIdx.x;
  if (i < n) p[i] = 0.0f;
}
__global__ void k_zero_u16(u16* __restrict__ p, int n) {
  int i = blockIdx.x * blockDim.x + threadIdx.x;
  if (i < n) p[i] = 0;
}
// dec_in[:, 0:VC] = one-hot at VC-1
__global__ void k_init_onehot(u16* __restrict__ dec_in) {
  int i = blockIdx.x * blockDim.x + threadIdx.x;   // B*VC
  int b = i / VC_, j = i % VC_;
  dec_in[(size_t)b * KDEC_ + j] = (j == VC_ - 1) ? (u16)0x3F80 : (u16)0;
}

// GRU gate: Gx = x@Wih^T + bih, Gh = h@Whh^T + bhh (biases already added by gemm)
__global__ void k_gru_gate(const float* __restrict__ Gx, const float* __restrict__ Gh,
                           const float* __restrict__ hprev,
                           float* __restrict__ hf, u16* __restrict__ hb, int bstride)
{
  int i = blockIdx.x * blockDim.x + threadIdx.x;   // B*H
  int b = i / H_, j = i % H_;
  size_t g = (size_t)b * H3_ + j;
  float r = sigm(Gx[g] + Gh[g]);
  float z = sigm(Gx[g + H_] + Gh[g + H_]);
  float n = tanhf(Gx[g + 2 * H_] + r * Gh[g + 2 * H_]);
  float h = (1.0f - z) * n + z * hprev[(size_t)b * H_ + j];
  hf[(size_t)b * H_ + j] = h;
  hb[(size_t)b * bstride + j] = f32_to_bf16(h);
}

// var = exp(vpre); z = mu + var*noise; scatter z (bf16) into decoder input
__global__ void k_latent(const float* __restrict__ mu, const float* __restrict__ vpre,
                         const float* __restrict__ noise,
                         float* __restrict__ var_out, float* __restrict__ z_out,
                         u16* __restrict__ zbf, u16* __restrict__ dec_in)
{
  int i = blockIdx.x * blockDim.x + threadIdx.x;   // B*Z
  int b = i / Z_, j = i % Z_;
  float v = expf(vpre[i]);
  float z = mu[i] + v * noise[i];
  var_out[i] = v;
  z_out[i]   = z;
  u16 zb = f32_to_bf16(z);
  zbf[i] = zb;
  dec_in[(size_t)b * KDEC_ + VC_ + j] = zb;
}

__global__ void k_tanh_init(const float* __restrict__ pre,
                            float* __restrict__ hf, u16* __restrict__ hb) {
  int i = blockIdx.x * blockDim.x + threadIdx.x;   // B*H
  float h = tanhf(pre[i]);
  hf[i] = h;
  hb[i] = f32_to_bf16(h);
}

// per-batch: log_softmax over C (stride-V groups), write recon[b,c,v,t],
// argmax over VC (first-occurrence tie-break), refresh one-hot feedback.
__global__ void k_lsm_argmax(const float* __restrict__ logits,
                             float* __restrict__ recon,
                             u16* __restrict__ dec_in, int t)
{
  __shared__ float sv[128];
  __shared__ int   sj[128];
  const int b = blockIdx.x;
  const int v = threadIdx.x;         // 0..127 (= V)
  const float* lb = logits + (size_t)b * VC_;

  float m = -3.4e38f;
#pragma unroll
  for (int c = 0; c < C_; ++c) m = fmaxf(m, lb[c * V_ + v]);
  float s = 0.0f;
#pragma unroll
  for (int c = 0; c < C_; ++c) s += expf(lb[c * V_ + v] - m);
  const float lse = m + logf(s);

  float best = -3.4e38f;
  int bestj = v;
  float* rb = recon + (size_t)b * C_ * V_ * NSTEP_ + t;
#pragma unroll
  for (int c = 0; c < C_; ++c) {
    int j = c * V_ + v;
    float ls = lb[j] - lse;
    rb[(size_t)j * NSTEP_] = ls;
    if (ls > best) { best = ls; bestj = j; }
  }
  sv[v] = best; sj[v] = bestj;
  __syncthreads();
  for (int off = 64; off > 0; off >>= 1) {
    if (v < off) {
      if (sv[v + off] > sv[v] || (sv[v + off] == sv[v] && sj[v + off] < sj[v])) {
        sv[v] = sv[v + off]; sj[v] = sj[v + off];
      }
    }
    __syncthreads();
  }
  u16* db = dec_in + (size_t)b * KDEC_;
  for (int j = v; j < VC_; j += 128) db[j] = 0;
  __syncthreads();
  if (v == 0) db[sj[0]] = 0x3F80;    // bf16 1.0
}

// ---------------- host orchestration ----------------
extern "C" void kernel_launch(void* const* d_in, const int* in_sizes, int n_in,
                              void* d_out, int out_size, void* d_ws, size_t ws_size,
                              hipStream_t stream) {
  (void)in_sizes; (void)n_in; (void)out_size; (void)ws_size;

  const float* x       = (const float*)d_in[0];
  const float* noise   = (const float*)d_in[1];
  const float* bih_f   = (const float*)d_in[4];
  const float* bhh_f   = (const float*)d_in[5];
  const float* bih_b   = (const float*)d_in[8];
  const float* bhh_b   = (const float*)d_in[9];
  const float* bmu     = (const float*)d_in[11];
  const float* bvar    = (const float*)d_in[13];
  const float* binit   = (const float*)d_in[15];
  const float* c1_bih  = (const float*)d_in[18];
  const float* c1_bhh  = (const float*)d_in[19];
  const float* c2_bih  = (const float*)d_in[22];
  const float* c2_bhh  = (const float*)d_in[23];
  const float* bout    = (const float*)d_in[25];

  float* out_mu    = (float*)d_out;
  float* out_var   = out_mu  + B_ * Z_;
  float* out_z     = out_var + B_ * Z_;
  float* out_recon = out_z   + B_ * Z_;

  // ---- workspace bump allocator ----
  char* base = (char*)d_ws;
  size_t off = 0;
  auto alloc = [&](size_t bytes) -> void* {
    off = (off + 255) & ~(size_t)255;
    void* p = base + off;
    off += bytes;
    return p;
  };
  auto allocU16 = [&](size_t n) { return (u16*)alloc(n * sizeof(u16)); };
  auto allocF32 = [&](size_t n) { return (float*)alloc(n * sizeof(float)); };

  // pre-transposed bf16 weights (K,N)
  u16* wt_f_Wih = allocU16((size_t)H3_ * V_);
  u16* wt_f_Whh = allocU16((size_t)H3_ * H_);
  u16* wt_b_Wih = allocU16((size_t)H3_ * V_);
  u16* wt_b_Whh = allocU16((size_t)H3_ * H_);
  u16* wt_Wmu   = allocU16((size_t)Z_ * 2 * H_);
  u16* wt_Wvar  = allocU16((size_t)Z_ * 2 * H_);
  u16* wt_Winit = allocU16((size_t)H_ * Z_);
  u16* wt_c1Wih = allocU16((size_t)H3_ * KDEC_);
  u16* wt_c1Whh = allocU16((size_t)H3_ * H_);
  u16* wt_c2Wih = allocU16((size_t)H3_ * H_);
  u16* wt_c2Whh = allocU16((size_t)H3_ * H_);
  u16* wt_Wout  = allocU16((size_t)VC_ * H_);
  u16* xt       = allocU16((size_t)S_ * B_ * V_);
  float* hf_f   = allocF32((size_t)B_ * H_);
  float* hb_f   = allocF32((size_t)B_ * H_);
  float* h1_f   = allocF32((size_t)B_ * H_);
  float* h2_f   = allocF32((size_t)B_ * H_);
  u16* hf_b     = allocU16((size_t)B_ * H_);
  u16* hb_b     = allocU16((size_t)B_ * H_);
  u16* h1_b     = allocU16((size_t)B_ * H_);
  u16* h2_b     = allocU16((size_t)B_ * H_);
  u16* enc      = allocU16((size_t)B_ * 2 * H_);
  u16* zbf      = allocU16((size_t)B_ * Z_);
  u16* dec_in   = allocU16((size_t)B_ * KDEC_);
  float* Gx     = allocF32((size_t)B_ * H3_);
  float* Gh     = allocF32((size_t)B_ * H3_);
  float* vpre   = allocF32((size_t)B_ * Z_);
  float* h1pre  = allocF32((size_t)B_ * H_);
  float* logits = allocF32((size_t)B_ * VC_);

  auto cvtT = [&](int idx, u16* dst, int N, int K) {
    int n = N * K;
    k_cvt_w_t<<<(n + 255) / 256, 256, 0, stream>>>((const float*)d_in[idx], dst, N, K);
  };
  auto gemm = [&](const u16* A, int lda, const u16* Wt,
                  const float* bias, float* Cp, int N, int K) {
    dim3 grid(N / BN, B_ / BM);
    k_gemm_bf16<<<grid, 256, 0, stream>>>(A, lda, Wt, bias, Cp, N, K);
  };
  auto gate = [&](const float* gx, const float* gh, const float* hprev,
                  float* hfp, u16* hbp, int bstride) {
    k_gru_gate<<<(B_ * H_) / 256, 256, 0, stream>>>(gx, gh, hprev, hfp, hbp, bstride);
  };

  // ---- one-time weight transpose+convert ----
  cvtT(2,  wt_f_Wih, H3_, V_);
  cvtT(3,  wt_f_Whh, H3_, H_);
  cvtT(6,  wt_b_Wih, H3_, V_);
  cvtT(7,  wt_b_Whh, H3_, H_);
  cvtT(10, wt_Wmu,   Z_, 2 * H_);
  cvtT(12, wt_Wvar,  Z_, 2 * H_);
  cvtT(14, wt_Winit, H_, Z_);
  cvtT(16, wt_c1Wih, H3_, KDEC_);
  cvtT(17, wt_c1Whh, H3_, H_);
  cvtT(20, wt_c2Wih, H3_, H_);
  cvtT(21, wt_c2Whh, H3_, H_);
  cvtT(24, wt_Wout,  VC_, H_);
  k_cvt_x<<<(S_ * B_ * V_) / 256, 256, 0, stream>>>(x, xt);

  // ---- state init ----
  k_zero_f32<<<(B_ * H_) / 256, 256, 0, stream>>>(hf_f, B_ * H_);
  k_zero_f32<<<(B_ * H_) / 256, 256, 0, stream>>>(hb_f, B_ * H_);
  k_zero_u16<<<(B_ * H_) / 256, 256, 0, stream>>>(hf_b, B_ * H_);
  k_zero_u16<<<(B_ * H_) / 256, 256, 0, stream>>>(hb_b, B_ * H_);
  k_init_onehot<<<(B_ * VC_) / 256, 256, 0, stream>>>(dec_in);

  // ---- bidirectional GRU encoder ----
  for (int t = 0; t < S_; ++t) {
    // forward direction
    gemm(xt + (size_t)t * B_ * V_, V_, wt_f_Wih, bih_f, Gx, H3_, V_);
    gemm(hf_b, H_, wt_f_Whh, bhh_f, Gh, H3_, H_);
    if (t == S_ - 1) gate(Gx, Gh, hf_f, hf_f, enc, 2 * H_);
    else             gate(Gx, Gh, hf_f, hf_f, hf_b, H_);
    // backward direction (reads reversed time)
    gemm(xt + (size_t)(S_ - 1 - t) * B_ * V_, V_, wt_b_Wih, bih_b, Gx, H3_, V_);
    gemm(hb_b, H_, wt_b_Whh, bhh_b, Gh, H3_, H_);
    if (t == S_ - 1) gate(Gx, Gh, hb_f, hb_f, enc + H_, 2 * H_);
    else             gate(Gx, Gh, hb_f, hb_f, hb_b, H_);
  }

  // ---- latent head ----
  gemm(enc, 2 * H_, wt_Wmu,  bmu,  out_mu, Z_, 2 * H_);
  gemm(enc, 2 * H_, wt_Wvar, bvar, vpre,   Z_, 2 * H_);
  k_latent<<<(B_ * Z_) / 256, 256, 0, stream>>>(out_mu, vpre, noise,
                                                out_var, out_z, zbf, dec_in);
  gemm(zbf, Z_, wt_Winit, binit, h1pre, H_, Z_);
  k_tanh_init<<<(B_ * H_) / 256, 256, 0, stream>>>(h1pre, h1_f, h1_b);

  // ---- autoregressive decoder ----
  for (int i = 0; i < NSTEP_; ++i) {
    gemm(dec_in, KDEC_, wt_c1Wih, c1_bih, Gx, H3_, KDEC_);
    gemm(h1_b, H_, wt_c1Whh, c1_bhh, Gh, H3_, H_);
    gate(Gx, Gh, h1_f, h1_f, h1_b, H_);          // h1 <- h1n

    const u16*   h2p_b = (i == 0) ? h1_b : h2_b;
    const float* h2p_f = (i == 0) ? h1_f : h2_f;
    gemm(h1_b, H_, wt_c2Wih, c2_bih, Gx, H3_, H_);
    gemm(h2p_b, H_, wt_c2Whh, c2_bhh, Gh, H3_, H_);
    gate(Gx, Gh, h2p_f, h2_f, h2_b, H_);         // h2 <- h2n

    gemm(h2_b, H_, wt_Wout, bout, logits, VC_, H_);
    k_lsm_argmax<<<B_, 128, 0, stream>>>(logits, out_recon, dec_in, i);
  }
}
